// RTRBM_4818953306654
// MI455X (gfx1250) — matrix operand order repaired and checked by reading, not compiled, not run on hardware
//
#include <hip/hip_runtime.h>
#include <hip/hip_bf16.h>
#include <stdint.h>

#define NVIS 4096
#define NHID 1024
#define TSTEPS 64
#define BATCH 32
#define SPLITK 16

typedef __bf16 bf16t;
typedef __attribute__((ext_vector_type(16))) __bf16 v16bf;
typedef __attribute__((ext_vector_type(8)))  __bf16 v8bf;
typedef __attribute__((ext_vector_type(8)))  float  v8f;

// ---------- helpers ----------

__device__ __forceinline__ bf16t f2bf(float f) {
    union { float f; unsigned u; } a; a.f = f;
    unsigned r = a.u + 0x7FFFu + ((a.u >> 16) & 1u);   // RNE
    unsigned short h = (unsigned short)(r >> 16);
    return __builtin_bit_cast(bf16t, h);
}

// Deterministic counter-based uniform in [0,1) (splitmix64 finalizer).
__device__ __forceinline__ float rng01(unsigned long long seed, unsigned int idx) {
    unsigned long long x = seed + 0x9E3779B97F4A7C15ull * (unsigned long long)(idx + 1u);
    x ^= x >> 30; x *= 0xBF58476D1CE4E5B9ull;
    x ^= x >> 27; x *= 0x94D049BB133111EBull;
    x ^= x >> 31;
    return (float)(unsigned int)(x >> 40) * (1.0f / 16777216.0f);
}

// ---------- one-time prep kernels ----------

__global__ void cvt_w_kernel(const float* __restrict__ W,
                             bf16t* __restrict__ W_bf, bf16t* __restrict__ Wt_bf) {
    long long idx = (long long)blockIdx.x * blockDim.x + threadIdx.x;
    if (idx >= (long long)NHID * NVIS) return;
    int r = (int)(idx / NVIS), c = (int)(idx % NVIS);
    float w = W[idx];
    W_bf[idx] = f2bf(w);
    Wt_bf[(long long)c * NHID + r] = f2bf(w);
}

__global__ void cvt_u_kernel(const float* __restrict__ U, bf16t* __restrict__ U_bf) {
    long long idx = (long long)blockIdx.x * blockDim.x + threadIdx.x;
    if (idx >= (long long)NHID * NHID) return;
    U_bf[idx] = f2bf(U[idx]);
}

// v [NVIS, T, B] f32  ->  vT [T][B][NVIS] bf16 (B-operand panels, row-contiguous in K)
__global__ void cvt_v_kernel(const float* __restrict__ v, bf16t* __restrict__ vT) {
    long long idx = (long long)blockIdx.x * blockDim.x + threadIdx.x;
    if (idx >= (long long)TSTEPS * BATCH * NVIS) return;
    int n = (int)(idx % NVIS);
    long long tb = idx / NVIS;
    int b = (int)(tb % BATCH);
    int t = (int)(tb / BATCH);
    vT[idx] = f2bf(v[(long long)n * (TSTEPS * BATCH) + (long long)t * BATCH + b]);
}

__global__ void bias_init_kernel(const float* __restrict__ b_init, float* __restrict__ hid_bias) {
    int idx = blockIdx.x * blockDim.x + threadIdx.x;
    if (idx >= NHID * 32) return;
    hid_bias[idx] = b_init[idx >> 5];
}

// ---------- split-K WMMA GEMM: partials Zp[slice][M][32] = A(M,K) @ Xt(32,K)^T ----------
// Block: 256 threads = 8 waves, 128 rows. Wave w: rows [bx*128 + (w>>1)*32, +32) as TWO
// 16-row tiles (independent accumulators, shared B fragment), cols [(w&1)*16, +16).
__global__ __launch_bounds__(256)
void gemm_partial_kernel(const bf16t* __restrict__ A, const bf16t* __restrict__ Xt,
                         float* __restrict__ Zp, int M, int K, int kchunk) {
    const int wave = threadIdx.x >> 5;
    const int lane = threadIdx.x & 31;
    const int hf   = lane >> 4;      // lane half (0: lanes 0-15, 1: lanes 16-31)
    const int ml   = lane & 15;
    const int row0 = blockIdx.x * 128 + (wave >> 1) * 32;   // tile0: row0.., tile1: row0+16..
    const int c0   = (wave & 1) * 16;
    const int kb   = blockIdx.y * kchunk;

    const bf16t* arow0 = A  + (size_t)(row0 + ml) * K + kb;        // A row, tile 0
    const bf16t* arow1 = A  + (size_t)(row0 + 16 + ml) * K + kb;   // A row, tile 1
    const bf16t* xrow  = Xt + (size_t)(c0 + ml) * K + kb;          // B column (transposed panel)

    v8f acc0 = {}, acc1 = {};
    #pragma unroll 2
    for (int k = 0; k < kchunk; k += 32) {
        // A fragment 16x32: elems[0:8] = K[hf*8 .. +8), elems[8:16] = K[16+hf*8 .. +8)
        v8bf a0lo = *(const v8bf*)(arow0 + k + hf * 8);
        v8bf a0hi = *(const v8bf*)(arow0 + k + 16 + hf * 8);
        v8bf a1lo = *(const v8bf*)(arow1 + k + hf * 8);
        v8bf a1hi = *(const v8bf*)(arow1 + k + 16 + hf * 8);
        // B fragment 32x16: elems[0:8] = K[hf*8 .. +8), elems[8:16] = K[16+hf*8 .. +8) of column c0+n
        v8bf blo = *(const v8bf*)(xrow + k + hf * 8);
        v8bf bhi = *(const v8bf*)(xrow + k + 16 + hf * 8);
        v16bf b = __builtin_shufflevector(blo, bhi, 0,1,2,3,4,5,6,7,8,9,10,11,12,13,14,15);
        __builtin_prefetch(arow0 + k + 256, 0, 3);   // global_prefetch_b8 on streamed W panel
        __builtin_prefetch(arow1 + k + 256, 0, 3);
        v16bf a0 = __builtin_shufflevector(a0lo, a0hi,
                                           0,1,2,3,4,5,6,7,8,9,10,11,12,13,14,15);
        v16bf a1 = __builtin_shufflevector(a1lo, a1hi,
                                           0,1,2,3,4,5,6,7,8,9,10,11,12,13,14,15);
        // two independent accumulation chains -> XDL latency overlap
        acc0 = __builtin_amdgcn_wmma_f32_16x16x32_bf16(
                   false, a0, false, b, (short)0, acc0, false, false);
        acc1 = __builtin_amdgcn_wmma_f32_16x16x32_bf16(
                   false, a1, false, b, (short)0, acc1, false, false);
    }

    // D layout: element i -> row = rowbase + 8*hf + i, col = c0 + (lane&15)
    float* zp = Zp + (size_t)blockIdx.y * M * 32;
    const int bcol = c0 + ml;
    #pragma unroll
    for (int i = 0; i < 8; ++i) {
        int r0 = row0 + hf * 8 + i;
        zp[(size_t)r0 * 32 + bcol] = acc0[i];
        int r1 = row0 + 16 + hf * 8 + i;
        zp[(size_t)r1 * 32 + bcol] = acc1[i];
    }
}

// ---------- fused epilogue: reduce split-K, bias, sigmoid, Bernoulli, scatter ----------
__global__ void epilogue_kernel(const float* __restrict__ Zp, int M,
                                const float* __restrict__ bias_vec,   // [M] or null
                                const float* __restrict__ bias_mat,   // [M*32] or null
                                float* lin_out,                       // linear mode: write z, stop
                                float* prob_strided,                  // [M,T,B] slice at t
                                bf16t* probT_bf,                      // [32,M] transposed bf16
                                float* samp_strided,                  // [M,T,B] slice at t
                                bf16t* sampT_bf,                      // [32,M] transposed bf16
                                int t, unsigned long long seed) {
    int idx = blockIdx.x * blockDim.x + threadIdx.x;
    if (idx >= M * 32) return;
    int r = idx >> 5, b = idx & 31;

    float z = 0.f;
    #pragma unroll
    for (int s = 0; s < SPLITK; ++s) z += Zp[(size_t)s * M * 32 + idx];
    z += bias_vec ? bias_vec[r] : bias_mat[idx];

    if (lin_out) { lin_out[idx] = z; return; }

    float p = 1.f / (1.f + __expf(-z));
    size_t so = (size_t)r * (TSTEPS * BATCH) + (size_t)t * BATCH + b;
    if (prob_strided) prob_strided[so] = p;
    if (probT_bf)     probT_bf[(size_t)b * M + r] = f2bf(p);
    float s01 = (rng01(seed, (unsigned)idx) < p) ? 1.f : 0.f;
    if (samp_strided) samp_strided[so] = s01;
    if (sampT_bf)     sampT_bf[(size_t)b * M + r] = f2bf(s01);
}

// ---------- host orchestration ----------

extern "C" void kernel_launch(void* const* d_in, const int* in_sizes, int n_in,
                              void* d_out, int out_size, void* d_ws, size_t ws_size,
                              hipStream_t stream) {
    (void)in_sizes; (void)n_in; (void)out_size; (void)ws_size;
    const float* v      = (const float*)d_in[0];
    const float* W      = (const float*)d_in[1];
    const float* U      = (const float*)d_in[2];
    const float* b_v    = (const float*)d_in[3];
    const float* b_h    = (const float*)d_in[4];
    const float* b_init = (const float*)d_in[5];
    // d_in[6] = CDk (== 2, chain hardcoded below)

    float* out_vm = (float*)d_out;                              // v_model [NV,T,B]
    float* out_rm = out_vm + (size_t)NVIS * TSTEPS * BATCH;     // r_model [NH,T,B]
    float* out_rt = out_rm + (size_t)NHID * TSTEPS * BATCH;     // rt      [NH,T,B]

    char* ws = (char*)d_ws;
    auto carve = [&](size_t bytes) -> char* {
        char* p = ws; ws += (bytes + 255) & ~(size_t)255; return p;
    };
    bf16t* W_bf   = (bf16t*)carve((size_t)NHID * NVIS * 2);
    bf16t* Wt_bf  = (bf16t*)carve((size_t)NHID * NVIS * 2);
    bf16t* U_bf   = (bf16t*)carve((size_t)NHID * NHID * 2);
    bf16t* vT_bf  = (bf16t*)carve((size_t)TSTEPS * BATCH * NVIS * 2);
    bf16t* rT_bf  = (bf16t*)carve((size_t)BATCH * NHID * 2);
    bf16t* hT_bf  = (bf16t*)carve((size_t)BATCH * NHID * 2);
    bf16t* vmT_bf = (bf16t*)carve((size_t)BATCH * NVIS * 2);
    float* hid_bias = (float*)carve((size_t)NHID * 32 * 4);
    float* Zp       = (float*)carve((size_t)SPLITK * NVIS * 32 * 4);

    { long long n = (long long)NHID * NVIS;
      cvt_w_kernel<<<(unsigned)((n + 255) / 256), 256, 0, stream>>>(W, W_bf, Wt_bf); }
    { long long n = (long long)NHID * NHID;
      cvt_u_kernel<<<(unsigned)((n + 255) / 256), 256, 0, stream>>>(U, U_bf); }
    { long long n = (long long)TSTEPS * BATCH * NVIS;
      cvt_v_kernel<<<(unsigned)((n + 255) / 256), 256, 0, stream>>>(v, vT_bf); }

    auto gemm = [&](const bf16t* A, const bf16t* Xt, int M, int K,
                    const float* bias_vec, const float* bias_mat,
                    float* lin_out, float* prob_s, bf16t* probT,
                    float* samp_s, bf16t* sampT, int t, unsigned long long seed) {
        dim3 grid(M / 128, SPLITK);
        gemm_partial_kernel<<<grid, 256, 0, stream>>>(A, Xt, Zp, M, K, K / SPLITK);
        int n = M * 32;
        epilogue_kernel<<<(n + 255) / 256, 256, 0, stream>>>(
            Zp, M, bias_vec, bias_mat, lin_out, prob_s, probT, samp_s, sampT, t, seed);
    };

    const unsigned long long base_seed = 0x5EED5EEDCAFEF00Dull;
    for (int t = 0; t < TSTEPS; ++t) {
        if (t == 0) {
            int n = NHID * 32;
            bias_init_kernel<<<(n + 255) / 256, 256, 0, stream>>>(b_init, hid_bias);
        } else {
            // hid_bias = U @ r_lag + b_h   (linear epilogue)
            gemm(U_bf, rT_bf, NHID, NHID, b_h, nullptr,
                 hid_bias, nullptr, nullptr, nullptr, nullptr, t, 0);
        }
        unsigned long long s = base_seed + (unsigned long long)t * 16ull;
        // r = sigmoid(W @ v_t + hid_bias): rt<-p, rT_bf<-p (recurrence), h1 = bern -> hT_bf
        gemm(W_bf, vT_bf + (size_t)t * BATCH * NVIS, NHID, NVIS, nullptr, hid_bias,
             nullptr, out_rt, rT_bf, nullptr, hT_bf, t, s + 0);
        // vm = bern(sigmoid(Wt @ h1 + b_v)) -> vmT_bf
        gemm(Wt_bf, hT_bf, NVIS, NHID, b_v, nullptr,
             nullptr, nullptr, nullptr, nullptr, vmT_bf, t, s + 1);
        // h2 = bern(sigmoid(W @ vm + hid_bias)) -> hT_bf, r_model<-sample
        gemm(W_bf, vmT_bf, NHID, NVIS, nullptr, hid_bias,
             nullptr, nullptr, nullptr, out_rm, hT_bf, t, s + 2);
        // v_model = bern(sigmoid(Wt @ h2 + b_v)) -> out_vm
        gemm(Wt_bf, hT_bf, NVIS, NHID, b_v, nullptr,
             nullptr, nullptr, nullptr, out_vm, nullptr, t, s + 3);
    }
}
